// Agent_55602646614622
// MI455X (gfx1250) — matrix operand order repaired and checked by reading, not compiled, not run on hardware
//
#include <hip/hip_runtime.h>
#include <hip/hip_bf16.h>
#include <stdint.h>

typedef __attribute__((ext_vector_type(8)))  __bf16 v8bf;
typedef __attribute__((ext_vector_type(16))) __bf16 v16bf;
typedef __attribute__((ext_vector_type(8)))  float  v8f;

#define BB 2048
#define AA 256

__device__ __forceinline__ float sigmoidf_(float x) { return 1.0f / (1.0f + __expf(-x)); }

// ---------------------------------------------------------------------------
// Weight conversion fp32 -> bf16; also fuses [W_ih | W_hh] into Wcat[1024,512]
// ---------------------------------------------------------------------------
__global__ void k_convw(const float* __restrict__ W_ih, const float* __restrict__ W_hh,
                        const float* __restrict__ W1,   const float* __restrict__ W2,
                        __bf16* __restrict__ Wcat, __bf16* __restrict__ W1b, __bf16* __restrict__ W2b) {
  int idx = blockIdx.x * 256 + threadIdx.x;
  if (idx < 1024 * 512) {
    int n = idx >> 9, k = idx & 511;
    float v = (k < 256) ? W_ih[n * 256 + k] : W_hh[n * 256 + (k - 256)];
    Wcat[idx] = (__bf16)v;
  } else if (idx < 1024 * 512 + 256 * 512) {
    int j = idx - 1024 * 512;
    W1b[j] = (__bf16)W1[j];
  } else if (idx < 1024 * 512 + 256 * 512 + 256 * 256) {
    int j = idx - (1024 * 512 + 256 * 512);
    W2b[j] = (__bf16)W2[j];
  }
}

// ---------------------------------------------------------------------------
// Build LSTM input Xcat = [rel_emb[prev] | ent_emb[cur] | h0] (bf16, [B,512])
// and sq tail = [ent_emb[cur] | rel_emb[query]] (f32, [B,256])
// ---------------------------------------------------------------------------
__global__ void k_prep(const int* __restrict__ prev_rel, const int* __restrict__ cur_ent,
                       const int* __restrict__ query_rel, const float* __restrict__ h0,
                       const float* __restrict__ rel_emb, const float* __restrict__ ent_emb,
                       __bf16* __restrict__ Xcat, float* __restrict__ tail) {
  int b = blockIdx.x, t = threadIdx.x;
  int pr = prev_rel[b], ce = cur_ent[b], qr = query_rel[b];
  float xv = (t < 128) ? rel_emb[(size_t)pr * 128 + t] : ent_emb[(size_t)ce * 128 + (t - 128)];
  Xcat[(size_t)b * 512 + t]       = (__bf16)xv;
  Xcat[(size_t)b * 512 + 256 + t] = (__bf16)h0[(size_t)b * 256 + t];
  tail[(size_t)b * 256 + t] = (t < 128) ? ent_emb[(size_t)ce * 128 + t]
                                        : rel_emb[(size_t)qr * 128 + (t - 128)];
}

// ---------------------------------------------------------------------------
// Generic bf16 WMMA GEMM: Z[B,N] = X[B,K] @ W[N,K]^T (+bias0 +bias1), f32 out.
// Block = 256 thr (8 waves); each wave -> one 16x16 tile. Grid (B/16, N/128).
// A frag: lane=M, K chunks [8*hi,+8) & [+16,+24). B frag: lane=N, same K chunks.
// ---------------------------------------------------------------------------
__global__ void k_gemm_bf16(const __bf16* __restrict__ X, const __bf16* __restrict__ W,
                            const float* __restrict__ bias0, const float* __restrict__ bias1,
                            float* __restrict__ Z, int K, int N) {
  const int lane = threadIdx.x & 31;
  const int wave = threadIdx.x >> 5;
  const int row0 = blockIdx.x * 16;
  const int n0   = blockIdx.y * 128 + wave * 16;
  const int m    = lane & 15;
  const int hi   = lane >> 4;
  const int kA   = hi * 8;
  const int kB   = hi * 16;
  const __bf16* Xrow = X + (size_t)(row0 + m) * K;
  const __bf16* Wrow = W + (size_t)(n0 + m) * K;   // lane&15 selects output column n
  v8f acc = {};
  for (int k0 = 0; k0 < K; k0 += 32) {
    if (k0 + 32 < K) __builtin_prefetch(Wrow + k0 + 32 + kB, 0, 3);
    v8bf alo = *(const v8bf*)(Xrow + k0 + kA);
    v8bf ahi = *(const v8bf*)(Xrow + k0 + kA + 16);
    v16bf afrag;
#pragma unroll
    for (int e = 0; e < 8; ++e) { afrag[e] = alo[e]; afrag[8 + e] = ahi[e]; }
    v8bf blo = *(const v8bf*)(Wrow + k0 + kA);
    v8bf bhi = *(const v8bf*)(Wrow + k0 + kA + 16);
    v16bf bfrag;
#pragma unroll
    for (int e = 0; e < 8; ++e) { bfrag[e] = blo[e]; bfrag[8 + e] = bhi[e]; }
    acc = __builtin_amdgcn_wmma_f32_16x16x32_bf16(false, afrag, false, bfrag,
                                                  (short)0, acc, false, false);
  }
  const int n = n0 + (lane & 15);
  float bias = 0.0f;
  if (bias0) bias += bias0[n];
  if (bias1) bias += bias1[n];
#pragma unroll
  for (int r = 0; r < 8; ++r) {
    // C/D layout: VGPR r -> M = r + 8*hi, N = lane&15
    Z[(size_t)(row0 + r + 8 * hi) * N + n] = acc[r] + bias;
  }
}

// ---------------------------------------------------------------------------
// LSTM elementwise over gates[B,1024] (gate order i,f,g,o); writes h,c and
// sq = [bf16(h) | bf16(tail)]
// ---------------------------------------------------------------------------
__global__ void k_lstm(const float* __restrict__ gates, const float* __restrict__ c0,
                       const float* __restrict__ tail, float* __restrict__ h_out,
                       float* __restrict__ c_out, __bf16* __restrict__ sq) {
  int b = blockIdx.x, j = threadIdx.x;
  const float* g = gates + (size_t)b * 1024;
  float gi = g[j], gf = g[256 + j], gg = g[512 + j], go = g[768 + j];
  float c = sigmoidf_(gf) * c0[(size_t)b * 256 + j] + sigmoidf_(gi) * tanhf(gg);
  float h = sigmoidf_(go) * tanhf(c);
  h_out[(size_t)b * 256 + j] = h;
  c_out[(size_t)b * 256 + j] = c;
  sq[(size_t)b * 512 + j]       = (__bf16)h;
  sq[(size_t)b * 512 + 256 + j] = (__bf16)tail[(size_t)b * 256 + j];
}

// ---------------------------------------------------------------------------
// BatchNorm (train): deterministic two-stage column sums over B=2048 rows
// ---------------------------------------------------------------------------
__global__ void k_colpart(const float* __restrict__ Z, float* __restrict__ psum,
                          float* __restrict__ psumsq) {
  int blk = blockIdx.x, j = threadIdx.x;
  float s = 0.f, ss = 0.f;
  int r0 = blk * 64;
  for (int r = 0; r < 64; ++r) { float v = Z[(size_t)(r0 + r) * 256 + j]; s += v; ss += v * v; }
  psum[blk * 256 + j] = s; psumsq[blk * 256 + j] = ss;
}

__global__ void k_bnfinal(const float* __restrict__ psum, const float* __restrict__ psumsq,
                          const float* __restrict__ g, const float* __restrict__ beta,
                          float* __restrict__ Ac, float* __restrict__ Bc) {
  int j = threadIdx.x;
  float s = 0.f, ss = 0.f;
  for (int k = 0; k < 32; ++k) { s += psum[k * 256 + j]; ss += psumsq[k * 256 + j]; }
  float mu  = s * (1.0f / 2048.0f);
  float var = ss * (1.0f / 2048.0f) - mu * mu;
  float inv = rsqrtf(var + 1e-5f);
  Ac[j] = g[j] * inv;
  Bc[j] = beta[j] - g[j] * mu * inv;
}

__global__ void k_act(const float* __restrict__ Z, const float* __restrict__ Ac,
                      const float* __restrict__ Bc, __bf16* __restrict__ bf_out,
                      float* __restrict__ f_out) {
  size_t idx = (size_t)blockIdx.x * 256 + threadIdx.x;
  int j = threadIdx.x;   // N=256 == blockDim, so col == tid
  float y = fmaxf(0.0f, Ac[j] * Z[idx] + Bc[j]);
  if (bf_out) bf_out[idx] = (__bf16)y;
  if (f_out)  f_out[idx]  = y;
}

// ---------------------------------------------------------------------------
// Fused scoring: gather-dot (L2-resident embedding tables) + mask + Gumbel
// argmax sample + log_softmax + loss + chosen_relations.  1 block per batch row.
// The out[b] row (1 KB) is broadcast into LDS via GLOBAL_LOAD_ASYNC_TO_LDS_B128
// (ASYNCcnt-tracked, no VGPR round trip), overlapping the index-gather loads.
// ---------------------------------------------------------------------------
__global__ void k_score(const int* __restrict__ nr, const int* __restrict__ ne,
                        const float* __restrict__ rel_emb, const float* __restrict__ ent_emb,
                        const float* __restrict__ outf, float* __restrict__ loss,
                        float* __restrict__ lp, float* __restrict__ act_out,
                        float* __restrict__ chosen_out) {
  __shared__ float ov[256];
  __shared__ float sc[256];
  __shared__ float red[256];
  __shared__ int   ridx[256];
  __shared__ int   s_act;
  const int b = blockIdx.x;
  const int a = threadIdx.x;
  // Async copy: 64 lanes (waves 0-1) x 16B = 1024B = out[b] row into LDS.
  // Flat->LDS address: low 32 bits of the generic pointer are the LDS address.
  if (a < 64) {
    unsigned lds_dst = (unsigned)(uintptr_t)(void*)ov + (unsigned)a * 16u;
    unsigned long long gsrc = (unsigned long long)(uintptr_t)(outf + (size_t)b * 256) +
                              (unsigned long long)a * 16ull;
    asm volatile("global_load_async_to_lds_b128 %0, %1, off"
                 :: "v"(lds_dst), "v"(gsrc) : "memory");
  }
  // Issue the per-thread index gathers while the async copy is in flight.
  int r = nr[(size_t)b * AA + a];
  int e = ne[(size_t)b * AA + a];
  asm volatile("s_wait_asynccnt 0x0" ::: "memory");
  __syncthreads();
  const float4* rp = (const float4*)(rel_emb + (size_t)r * 128);
  const float4* ep = (const float4*)(ent_emb + (size_t)e * 128);
  const float4* o0 = (const float4*)(ov);
  const float4* o1 = (const float4*)(ov + 128);
  float s = 0.f;
#pragma unroll 4
  for (int i = 0; i < 32; ++i) {
    float4 rv = rp[i], o = o0[i];
    s += rv.x * o.x + rv.y * o.y + rv.z * o.z + rv.w * o.w;
  }
#pragma unroll 4
  for (int i = 0; i < 32; ++i) {
    float4 ev = ep[i], o = o1[i];
    s += ev.x * o.x + ev.y * o.y + ev.z * o.z + ev.w * o.w;
  }
  if (r == 0) s = -99999.0f;
  sc[a] = s;
  // Deterministic Gumbel noise (splitmix64 of (b,a) salted by key 42)
  uint64_t zz = (((uint64_t)(uint32_t)b << 32) | (uint32_t)a) ^ 0x2545F4914F6CDD1DULL;
  zz += 0x9E3779B97F4A7C15ULL;
  zz = (zz ^ (zz >> 30)) * 0xBF58476D1CE4E5B9ULL;
  zz = (zz ^ (zz >> 27)) * 0x94D049BB133111EBULL;
  zz ^= zz >> 31;
  float u = ((float)(uint32_t)(zz >> 40) + 0.5f) * (1.0f / 16777216.0f);
  red[a] = s + (-__logf(-__logf(u)));
  ridx[a] = a;
  __syncthreads();
  for (int off = 128; off > 0; off >>= 1) {
    if (a < off && red[a + off] > red[a]) { red[a] = red[a + off]; ridx[a] = ridx[a + off]; }
    __syncthreads();
  }
  if (a == 0) s_act = ridx[0];
  __syncthreads();
  red[a] = sc[a];
  __syncthreads();
  for (int off = 128; off > 0; off >>= 1) {
    if (a < off) red[a] = fmaxf(red[a], red[a + off]);
    __syncthreads();
  }
  float mx = red[0];
  __syncthreads();
  red[a] = __expf(sc[a] - mx);
  __syncthreads();
  for (int off = 128; off > 0; off >>= 1) {
    if (a < off) red[a] += red[a + off];
    __syncthreads();
  }
  float lse = mx + __logf(red[0]);
  lp[(size_t)b * AA + a] = sc[a] - lse;
  if (a == 0) {
    int act = s_act;
    loss[b]       = lse - sc[act];
    act_out[b]    = (float)act;
    chosen_out[b] = (float)nr[(size_t)b * AA + act];
  }
}

// ---------------------------------------------------------------------------
extern "C" void kernel_launch(void* const* d_in, const int* in_sizes, int n_in,
                              void* d_out, int out_size, void* d_ws, size_t ws_size,
                              hipStream_t stream) {
  (void)in_sizes; (void)n_in; (void)out_size; (void)ws_size;
  const int*   nrel    = (const int*)d_in[0];
  const int*   nent    = (const int*)d_in[1];
  const int*   cent    = (const int*)d_in[2];
  const int*   prel    = (const int*)d_in[3];
  const int*   qrel    = (const int*)d_in[4];
  const float* h0      = (const float*)d_in[6];
  const float* c0      = (const float*)d_in[7];
  const float* rel_emb = (const float*)d_in[8];
  const float* ent_emb = (const float*)d_in[9];
  const float* W_ih    = (const float*)d_in[10];
  const float* W_hh    = (const float*)d_in[11];
  const float* b_ih    = (const float*)d_in[12];
  const float* b_hh    = (const float*)d_in[13];
  const float* W1      = (const float*)d_in[14];
  const float* b1      = (const float*)d_in[15];
  const float* g1      = (const float*)d_in[16];
  const float* beta1   = (const float*)d_in[17];
  const float* W2      = (const float*)d_in[18];
  const float* b2      = (const float*)d_in[19];
  const float* g2      = (const float*)d_in[20];
  const float* beta2   = (const float*)d_in[21];

  char* ws = (char*)d_ws;
  size_t off = 0;
  auto carve = [&](size_t bytes) -> void* {
    void* p = ws + off;
    off = (off + bytes + 255) & ~(size_t)255;
    return p;
  };
  __bf16* Xcat   = (__bf16*)carve(2048ull * 512 * 2);
  __bf16* Wcat   = (__bf16*)carve(1024ull * 512 * 2);
  __bf16* W1b    = (__bf16*)carve(256ull * 512 * 2);
  __bf16* W2b    = (__bf16*)carve(256ull * 256 * 2);
  float*  tail   = (float*)carve(2048ull * 256 * 4);
  float*  gates  = (float*)carve(2048ull * 1024 * 4);   // 8 MB, reused below
  __bf16* sq     = (__bf16*)carve(2048ull * 512 * 2);
  float*  psum1  = (float*)carve(32ull * 256 * 4);
  float*  psumsq1= (float*)carve(32ull * 256 * 4);
  float*  psum2  = (float*)carve(32ull * 256 * 4);
  float*  psumsq2= (float*)carve(32ull * 256 * 4);
  float*  Ac1    = (float*)carve(256 * 4);
  float*  Bc1    = (float*)carve(256 * 4);
  float*  Ac2    = (float*)carve(256 * 4);
  float*  Bc2    = (float*)carve(256 * 4);
  // Reuse the gates region once k_lstm has consumed it:
  float*  z1   = gates;                                            // [0,2MB)
  __bf16* a1   = (__bf16*)((char*)gates + 2048ull * 256 * 4);      // [2MB,3MB)
  float*  z2   = (float*)((char*)a1 + 2048ull * 256 * 2);          // [3MB,5MB)
  float*  outf = (float*)((char*)z2 + 2048ull * 256 * 4);          // [5MB,7MB)

  float* out      = (float*)d_out;
  float* o_loss   = out;
  float* o_lp     = o_loss + 2048;
  float* o_h      = o_lp + 2048ull * 256;
  float* o_c      = o_h + 2048ull * 256;
  float* o_act    = o_c + 2048ull * 256;
  float* o_chosen = o_act + 2048;

  k_convw<<<2816, 256, 0, stream>>>(W_ih, W_hh, W1, W2, Wcat, W1b, W2b);
  k_prep<<<2048, 256, 0, stream>>>(prel, cent, qrel, h0, rel_emb, ent_emb, Xcat, tail);
  k_gemm_bf16<<<dim3(128, 8), 256, 0, stream>>>(Xcat, Wcat, b_ih, b_hh, gates, 512, 1024);
  k_lstm<<<2048, 256, 0, stream>>>(gates, c0, tail, o_h, o_c, sq);
  k_gemm_bf16<<<dim3(128, 2), 256, 0, stream>>>(sq, W1b, b1, nullptr, z1, 512, 256);
  k_colpart<<<32, 256, 0, stream>>>(z1, psum1, psumsq1);
  k_bnfinal<<<1, 256, 0, stream>>>(psum1, psumsq1, g1, beta1, Ac1, Bc1);
  k_act<<<2048, 256, 0, stream>>>(z1, Ac1, Bc1, a1, nullptr);
  k_gemm_bf16<<<dim3(128, 2), 256, 0, stream>>>(a1, W2b, b2, nullptr, z2, 256, 256);
  k_colpart<<<32, 256, 0, stream>>>(z2, psum2, psumsq2);
  k_bnfinal<<<1, 256, 0, stream>>>(psum2, psumsq2, g2, beta2, Ac2, Bc2);
  k_act<<<2048, 256, 0, stream>>>(z2, Ac2, Bc2, nullptr, outf);
  k_score<<<2048, 256, 0, stream>>>(nrel, nent, rel_emb, ent_emb, outf,
                                    o_loss, o_lp, o_act, o_chosen);
}
